// Decoder_53506702574128
// MI455X (gfx1250) — compile-verified, matrix operand order checked by reading, not run on hardware
//
#include <hip/hip_runtime.h>
#include <cstdint>
#include <cstddef>

// ---------------------------------------------------------------- config
#ifndef USE_ASYNC_LDS
#define USE_ASYNC_LDS 1   // global_load_async_to_lds_b64 staging (flip to 0 to fall back)
#endif

// ---------------------------------------------------------------- types
typedef _Float16 half_t;
typedef __attribute__((ext_vector_type(2)))  _Float16 v2h;
typedef __attribute__((ext_vector_type(4)))  _Float16 v4h;
typedef __attribute__((ext_vector_type(16))) _Float16 v16h;
typedef __attribute__((ext_vector_type(8)))  float    v8f;

// ---------------------------------------------------------------- dims
constexpr int B_      = 8;
constexpr int N_      = 128;
constexpr int KNEI    = 9;
constexpr int HID     = 256;
constexpr int VOCAB   = 21;
constexpr int NODE_IN = 6;
constexpr int POS_DIM = 16;
constexpr int NUM_RBF = 16;
constexpr int EDGE_IN = POS_DIM + NUM_RBF;   // 32
constexpr int DEPTH   = 4;
constexpr float AUG_EPS = 0.1f;

constexpr int KPAD_WV  = 288;   // 262 -> 288
constexpr int KPAD_MSG = 512;
constexpr int KPAD_NEI = 544;   // 528 -> 544
constexpr int NT       = HID / 16;  // 16 column tiles

// acc layout: 0 vnum 1 vden 2 nnum 3 nden 4 dnum 5 dden 6 snum 7 sden
#define ACC_V 0
#define ACC_N 2
#define ACC_D 4
#define ACC_S 6

// ---------------------------------------------------------------- device helpers
static __device__ __forceinline__ v8f wmma16(v16h a, v16h b, v8f c) {
  return __builtin_amdgcn_wmma_f32_16x16x32_f16(
      /*neg_a*/false, a, /*neg_b*/false, b,
      /*c_mod*/(short)0, c, /*reuse_a*/false, /*reuse_b*/false);
}

// A fragment: 16-bit A matrix 16x32; lanes 0-15 M=0..15 K 0-7,16-23, lanes 16-31 K 8-15,24-31.
#define LOAD_AFRAG(dstA, tile, lane)                                        \
  {                                                                         \
    const int m_  = (lane) & 15;                                            \
    const int kh_ = ((lane) >> 4) * 8;                                      \
    _Pragma("unroll")                                                       \
    for (int v_ = 0; v_ < 8; ++v_) {                                        \
      const int kb_ = (v_ < 4) ? (v_ * 2) : (16 + (v_ - 4) * 2);            \
      v2h p_ = *reinterpret_cast<const v2h*>(&(tile)[m_ * 32 + kb_ + kh_]); \
      (dstA)[2 * v_]     = p_[0];                                           \
      (dstA)[2 * v_ + 1] = p_[1];                                           \
    }                                                                       \
  }

static __device__ __forceinline__ v16h load_bfrag(const half_t* Bf,
                                                  int chunkIdx, int lane) {
  return *reinterpret_cast<const v16h*>(Bf + ((size_t)chunkIdx * 32 + lane) * 16);
}

// 8-byte memory -> LDS stage. Async path uses the CDNA5 async-DMA lane
// (ASYNCcnt); LDS byte offset is the low 32 bits of the generic pointer
// (ISA 10.2: LDS aperture truncates to addr[31:0]).
static __device__ __forceinline__ void stage8(const half_t* src, half_t* ldsDst) {
#if USE_ASYNC_LDS
  unsigned      lo = (unsigned)(size_t)ldsDst;
  unsigned long long ga = (unsigned long long)(size_t)src;
  asm volatile("global_load_async_to_lds_b64 %0, %1, off" :: "v"(lo), "v"(ga) : "memory");
#else
  *reinterpret_cast<v4h*>(ldsDst) = *reinterpret_cast<const v4h*>(src);
#endif
}
static __device__ __forceinline__ void stage_wait() {
#if USE_ASYNC_LDS
  asm volatile("s_wait_asynccnt 0x0" ::: "memory");
#endif
}

static __device__ __forceinline__ float pe_freq(int q) {
  // exp(-ln(10000)/16 * 2q), q = 0..7
  return __expf(-logf(10000.0f) * (float)(2 * q) / (float)POS_DIM);
}

static __device__ __forceinline__ void cross3(const float* a, const float* b, float* o) {
  o[0] = a[1] * b[2] - a[2] * b[1];
  o[1] = a[2] * b[0] - a[0] * b[2];
  o[2] = a[0] * b[1] - a[1] * b[0];
}
static __device__ __forceinline__ float dot3(const float* a, const float* b) {
  return a[0] * b[0] + a[1] * b[1] + a[2] * b[2];
}
static __device__ __forceinline__ void norm3(float* v) {
  float n = sqrtf(dot3(v, v)) + 1e-8f;
  v[0] /= n; v[1] /= n; v[2] /= n;
}

// ---------------------------------------------------------------- small kernels
__global__ void k_zero(float* p, int n) {
  int t = blockIdx.x * blockDim.x + threadIdx.x;
  if (t < n) p[t] = 0.f;
}

__global__ void k_axpy(const float* x, const float* nz, float* o, float a, int n) {
  int t = blockIdx.x * blockDim.x + threadIdx.x;
  if (t < n) o[t] = x[t] + a * nz[t];
}

// dihedral angles -> V (B,N,6)
__global__ void k_dihedral(const float* __restrict__ X4, float* __restrict__ V) {
  int t = blockIdx.x * blockDim.x + threadIdx.x;
  if (t >= B_ * 3 * N_) return;
  int b = t / (3 * N_), idx = t % (3 * N_);
  float cD = 1.f, sD = 0.f;
  if (idx >= 1 && idx <= 3 * N_ - 3) {
    int k = idx - 1;
    float P[4][3];
#pragma unroll
    for (int q = 0; q < 4; ++q) {
      int ta = k + q, nn = ta / 3, aa = ta % 3;
      for (int c = 0; c < 3; ++c)
        P[q][c] = X4[(((size_t)b * N_ + nn) * 4 + aa) * 3 + c];
    }
    float u2[3], u1[3], u0[3], n2[3], n1[3];
    for (int c = 0; c < 3; ++c) {
      u2[c] = P[1][c] - P[0][c];
      u1[c] = P[2][c] - P[1][c];
      u0[c] = P[3][c] - P[2][c];
    }
    norm3(u2); norm3(u1); norm3(u0);
    cross3(u2, u1, n2); norm3(n2);
    cross3(u1, u0, n1); norm3(n1);
    float cd = dot3(n2, n1);
    cd = fminf(fmaxf(cd, -1.f + 1e-7f), 1.f - 1e-7f);
    float sv = dot3(u2, n1);
    float sg = (sv > 0.f ? 1.f : 0.f) - (sv < 0.f ? 1.f : 0.f);
    float D = sg * acosf(cd);
    cD = cosf(D); sD = sinf(D);
  }
  int nn = idx / 3, cc = idx % 3;
  V[((size_t)b * N_ + nn) * 6 + cc]     = cD;
  V[((size_t)b * N_ + nn) * 6 + cc + 3] = sD;
}

// kNN over Xca (atom 1)
__global__ void k_knn(const float* __restrict__ X4, const float* __restrict__ mask,
                      int* __restrict__ Eidx, float* __restrict__ Dnei, int includeSelf) {
  int t = blockIdx.x * blockDim.x + threadIdx.x;
  if (t >= B_ * N_) return;
  int b = t / N_, i = t % N_;
  float mi = mask[b * N_ + i];
  float xi[3];
  for (int c = 0; c < 3; ++c) xi[c] = X4[(((size_t)b * N_ + i) * 4 + 1) * 3 + c];
  float d[N_];
  float rowmax = 0.f;
  for (int j = 0; j < N_; ++j) {
    float ss = 1e-6f;
    for (int c = 0; c < 3; ++c) {
      float df = xi[c] - X4[(((size_t)b * N_ + j) * 4 + 1) * 3 + c];
      ss += df * df;
    }
    float dist = sqrtf(ss) * (mi * mask[b * N_ + j]);
    d[j] = dist;
    rowmax = fmaxf(rowmax, dist);
  }
  unsigned chosen[4] = {0u, 0u, 0u, 0u};
  for (int kk = 0; kk < KNEI; ++kk) {
    float best = 3.4e38f; int bj = 0;
    for (int j = 0; j < N_; ++j) {
      if ((chosen[j >> 5] >> (j & 31)) & 1u) continue;
      float m2 = mi * mask[b * N_ + j];
      float tr = includeSelf ? (j <= i ? 1.f : 0.f) : (j < i ? 1.f : 0.f);
      float adj = d[j] + (1.f - m2 * tr) * (rowmax + 1.f);
      if (adj < best) { best = adj; bj = j; }
    }
    chosen[bj >> 5] |= 1u << (bj & 31);
    Eidx[(size_t)t * KNEI + kk] = bj;
    if (Dnei) Dnei[(size_t)t * KNEI + kk] = best;
  }
}

// edge features -> f16 (B*N*K, 32)
__global__ void k_edge(const int* __restrict__ Eidx, const float* __restrict__ Dnei,
                       half_t* __restrict__ E16) {
  int t = blockIdx.x * blockDim.x + threadIdx.x;
  if (t >= B_ * N_ * KNEI) return;
  int i = (t / KNEI) % N_;
  float di = (float)(Eidx[t] - i);
  half_t* o = E16 + (size_t)t * EDGE_IN;
#pragma unroll
  for (int q = 0; q < 8; ++q) {
    float ang = di * pe_freq(q);
    o[q]     = (half_t)cosf(ang);
    o[q + 8] = (half_t)sinf(ang);
  }
  float D = Dnei[t];
#pragma unroll
  for (int m = 0; m < NUM_RBF; ++m) {
    float mu = 20.f * (float)m / (float)(NUM_RBF - 1);
    float z = (D - mu) / (20.f / (float)NUM_RBF);
    o[POS_DIM + m] = (half_t)__expf(-z * z);
  }
}

// weight (K,Nout=256) fp32 -> WMMA B fragments f16, layout [(kc*NT+nc)*32+lane][16]
__global__ void k_swizzle(const float* __restrict__ W, half_t* __restrict__ out,
                          int K, int Kpad) {
  int t = blockIdx.x * blockDim.x + threadIdx.x;
  int total = (Kpad / 32) * NT * 32;
  if (t >= total) return;
  int lane = t & 31, rest = t >> 5;
  int nc = rest % NT, kc = rest / NT;
  int n  = nc * 16 + (lane & 15);
  int kh = (lane >> 4) * 8;
  half_t* dst = out + (size_t)t * 16;
#pragma unroll
  for (int v = 0; v < 8; ++v) {
    int kb = (v < 4) ? (v * 2) : (16 + (v - 4) * 2);
#pragma unroll
    for (int h = 0; h < 2; ++h) {
      int kk = kc * 32 + kb + h + kh;
      float val = (kk < K) ? W[(size_t)kk * HID + n] : 0.f;
      dst[2 * v + h] = (half_t)val;
    }
  }
}

// A for W_v: concat(V, W_s[S]) padded to 288, f16
__global__ void k_fill_av(const float* __restrict__ V, const float* __restrict__ Ws,
                          const int* __restrict__ S, half_t* __restrict__ Av) {
  int t = blockIdx.x * blockDim.x + threadIdx.x;
  if (t >= B_ * N_ * KPAD_WV) return;
  int r = t / KPAD_WV, k = t % KPAD_WV;
  float v = 0.f;
  if (k < NODE_IN)            v = V[(size_t)r * NODE_IN + k];
  else if (k < NODE_IN + HID) v = Ws[(size_t)S[r] * HID + (k - NODE_IN)];
  Av[(size_t)r * KPAD_WV + k] = (half_t)v;
}

// ---------------------------------------------------------------- generic WMMA GEMM
// C = relu(A[M,Kpad] @ Wfrag + bias) ; 256 thr = 8 waves -> 16 rows x 256 cols.
// Double-buffered LDS A tile, async memory->LDS staging, 2 WMMA / wave / chunk.
__global__ void k_gemm(const half_t* __restrict__ A, const half_t* __restrict__ Bf,
                       const float* __restrict__ bias, const float* __restrict__ rowmask,
                       float* __restrict__ Cf, half_t* __restrict__ Ch,
                       int M, int Kpad, int relu) {
  __shared__ half_t tile[2][16 * 32];
  const int tid  = threadIdx.x;
  const int lane = tid & 31;
  const int wave = tid >> 5;
  const int rowBase = blockIdx.x * 16;
  const int KC = Kpad / 32;
  const int nc0 = wave * 2, nc1 = nc0 + 1;

  // staging role: threads 0..127 move 8 bytes each (16 rows x 32 halves)
  const int srow = tid >> 3;
  const int soff = (tid & 7) * 4;
  const int gr   = min(rowBase + srow, M - 1);
  const half_t* srcBase = A + (size_t)gr * Kpad + soff;

  v8f acc0 = {0.f, 0.f, 0.f, 0.f, 0.f, 0.f, 0.f, 0.f};
  v8f acc1 = acc0;

  if (tid < 128) stage8(srcBase, &tile[0][srow * 32 + soff]);
  stage_wait();
  __syncthreads();

  for (int kc = 0; kc < KC; ++kc) {
    const int cur = kc & 1;
    if (kc + 1 < KC) {
      if (tid < 128)
        stage8(srcBase + (size_t)(kc + 1) * 32, &tile[cur ^ 1][srow * 32 + soff]);
      __builtin_prefetch(Bf + ((size_t)(kc + 1) * NT + nc0) * 32 * 16, 0, 0);
    }
    v16h a;
    LOAD_AFRAG(a, tile[cur], lane);
    v16h b0 = load_bfrag(Bf, kc * NT + nc0, lane);
    v16h b1 = load_bfrag(Bf, kc * NT + nc1, lane);
    acc0 = wmma16(a, b0, acc0);
    acc1 = wmma16(a, b1, acc1);
    stage_wait();
    __syncthreads();
  }

  const int kh = lane >> 4, n = lane & 15;
#pragma unroll
  for (int r8 = 0; r8 < 8; ++r8) {
    int m = r8 + kh * 8;
    int grow = rowBase + m;
    if (grow < M) {
      float rm = rowmask ? rowmask[grow] : 1.f;
#pragma unroll
      for (int c = 0; c < 2; ++c) {
        int col = (nc0 + c) * 16 + n;
        float v = (c ? acc1[r8] : acc0[r8]) + bias[col];
        if (relu) v = fmaxf(v, 0.f);
        v *= rm;
        if (Cf) Cf[(size_t)grow * HID + col] = v;
        if (Ch) Ch[(size_t)grow * HID + col] = (half_t)v;
      }
    }
  }
}

// A for W_h: concat(h[nei], h_e) f16, per edge, 512 wide
__global__ void k_fill_amsg(const half_t* __restrict__ h16, const half_t* __restrict__ he16,
                            const int* __restrict__ Eidx, half_t* __restrict__ Am) {
  int t = blockIdx.x * blockDim.x + threadIdx.x;
  if (t >= B_ * N_ * KNEI * 32) return;
  int e = t >> 5, q = t & 31;
  int b = e / (N_ * KNEI);
  int nei = b * N_ + Eidx[e];
  half_t* dst = Am + (size_t)e * 512 + q * 16;
  const half_t* src = (q < 16) ? (h16 + (size_t)nei * HID + q * 16)
                               : (he16 + (size_t)e * HID + (q - 16) * 16);
#pragma unroll
  for (int u = 0; u < 16; ++u) dst[u] = src[u];
}

// msg reduce over K edges + build A_U = concat(h, msg_sum)
__global__ void k_reduce(const half_t* __restrict__ msg, const int* __restrict__ Eidx,
                         const float* __restrict__ mask, const half_t* __restrict__ h16,
                         half_t* __restrict__ AU) {
  int t = blockIdx.x * blockDim.x + threadIdx.x;
  if (t >= B_ * N_ * HID) return;
  int r = t >> 8, c = t & 255;
  int b = r / N_;
  float s = 0.f;
#pragma unroll
  for (int kk = 0; kk < KNEI; ++kk) {
    int e = r * KNEI + kk;
    float nm = mask[b * N_ + Eidx[e]];
    s += (float)msg[(size_t)e * HID + c] * nm;
  }
  AU[(size_t)r * 512 + c]       = h16[(size_t)r * HID + c];
  AU[(size_t)r * 512 + HID + c] = (half_t)s;
}

// ---------------------------------------------------------------- losses
__global__ void k_vloss(const float* __restrict__ h, const float* __restrict__ W,
                        const float* __restrict__ bias, const float* __restrict__ tV,
                        const float* __restrict__ mask, float* __restrict__ acc) {
  int t = blockIdx.x * blockDim.x + threadIdx.x;
  if (t >= B_ * (N_ - 1)) return;
  int b = t / (N_ - 1), np = t % (N_ - 1);
  int row = b * N_ + np;
  float l = 0.f;
#pragma unroll
  for (int o = 0; o < NODE_IN; ++o) {
    float s = bias[o];
    for (int c = 0; c < HID; ++c) s += h[(size_t)row * HID + c] * W[(size_t)c * NODE_IN + o];
    float d = s - tV[((size_t)b * N_ + np + 1) * NODE_IN + o];
    l += d * d;
  }
  l /= (float)NODE_IN;
  float m = mask[b * N_ + np + 1];
  atomicAdd(&acc[ACC_V],     l * m);
  atomicAdd(&acc[ACC_V + 1], m);
}

__global__ void k_neilabel(const int* __restrict__ Etrue, const float* __restrict__ mask,
                           const float* __restrict__ X4, float* __restrict__ nlab,
                           float* __restrict__ nmask, float* __restrict__ dlab,
                           float* __restrict__ dmask) {
  int t = blockIdx.x * blockDim.x + threadIdx.x;
  if (t >= B_ * (N_ - 1) * N_) return;
  int j  = t % N_;
  int ip = (t / N_) % (N_ - 1);
  int b  = t / (N_ * (N_ - 1));
  int e  = ip - j;
  float valid = (e >= 0) ? 1.f : 0.f;
  int ec = min(max(e, 0), N_ - 1);
  float nm = valid * mask[b * N_ + ec] * mask[b * N_ + ip + 1];
  int lab = 0;
#pragma unroll
  for (int kk = 0; kk < KNEI; ++kk)
    if (Etrue[((size_t)b * N_ + ip + 1) * KNEI + kk] == ec) lab = 1;
  nlab[t]  = (float)lab * nm;
  nmask[t] = nm;
  if (j < KNEI) {
    float ss = 1e-6f;
    for (int c = 0; c < 3; ++c) {
      float df = X4[(((size_t)b * N_ + ip + 1) * 4 + 1) * 3 + c] -
                 X4[(((size_t)b * N_ + ec) * 4 + 1) * 3 + c];
      ss += df * df;
    }
    float dist = sqrtf(ss);
    float dl = (fminf(dist, 20.f) - 10.f) / 10.f;
    size_t i9 = ((size_t)b * (N_ - 1) + ip) * KNEI + j;
    dlab[i9]  = dl;
    dmask[i9] = nm;
  }
}

// ---------------------------------------------------------------- fused h_nei MLP (O_nei / O_dist)
// block 256 thr = 8 waves -> 16 rows x 256 cols of l1, then l2 dot + loss reduce.
// Double-buffered staged A tile (gathered h ++ analytic pos-embed), 1 barrier/chunk.
// mode 0: BCE-with-logits vs nlabel; mode 1: MSE vs dlab.
__global__ void k_bigmlp(const half_t* __restrict__ h16, const half_t* __restrict__ W1f,
                         const float* __restrict__ b1, const float* __restrict__ w2,
                         const float* __restrict__ b2, const float* __restrict__ lab,
                         const float* __restrict__ lmask, float* __restrict__ accN,
                         float* __restrict__ accD, int rows, int mode) {
  __shared__ half_t tile[2][16 * 32];
  __shared__ float  l1buf[16 * HID];
  __shared__ float  part[256];
  __shared__ float  rl[16], rd[16];

  const int tid  = threadIdx.x;
  const int lane = tid & 31;
  const int wave = tid >> 5;
  const int rowBase = blockIdx.x * 16;
  const int KC = KPAD_NEI / 32;

  // per-thread staging role (constant across K chunks)
  const int srow  = tid >> 4;
  const int spair = tid & 15;
  const int sr = rowBase + srow;
  const bool svalid = sr < rows;
  int sb = 0, sip = 0, sj = 0;
  if (svalid) {
    if (mode == 0) { sj = sr % N_; sip = (sr / N_) % (N_ - 1); sb = sr / (N_ * (N_ - 1)); }
    else           { sj = sr % KNEI; sip = (sr / KNEI) % (N_ - 1); sb = sr / (KNEI * (N_ - 1)); }
  }
  const int sec  = min(max(sip - sj, 0), N_ - 1);
  const int spos = (sip + 1) - sec;

  auto stage_chunk = [&](int kc, int buf) {
#pragma unroll
    for (int u = 0; u < 2; ++u) {
      int k = kc * 32 + spair * 2 + u;
      half_t v = (half_t)0;
      if (svalid) {
        if (k < HID) {
          v = h16[((size_t)sb * N_ + sip) * HID + k];
        } else if (k < 2 * HID) {
          v = h16[((size_t)sb * N_ + sec) * HID + (k - HID)];
        } else if (k < 2 * HID + POS_DIM) {
          int d = k - 2 * HID;
          float ang = (float)spos * pe_freq(d & 7);
          v = (half_t)((d < 8) ? cosf(ang) : sinf(ang));
        }
      }
      tile[buf][srow * 32 + spair * 2 + u] = v;
    }
  };

  v8f acc0 = {0.f, 0.f, 0.f, 0.f, 0.f, 0.f, 0.f, 0.f};
  v8f acc1 = acc0;

  stage_chunk(0, 0);
  __syncthreads();

  for (int kc = 0; kc < KC; ++kc) {
    const int cur = kc & 1;
    if (kc + 1 < KC) {
      stage_chunk(kc + 1, cur ^ 1);
      __builtin_prefetch(W1f + ((size_t)(kc + 1) * NT + wave * 2) * 32 * 16, 0, 0);
    }
    v16h a;
    LOAD_AFRAG(a, tile[cur], lane);
    v16h bf0 = load_bfrag(W1f, kc * NT + wave * 2,     lane);
    v16h bf1 = load_bfrag(W1f, kc * NT + wave * 2 + 1, lane);
    acc0 = wmma16(a, bf0, acc0);
    acc1 = wmma16(a, bf1, acc1);
    __syncthreads();
  }

  // bias + relu -> l1buf
  {
    const int kh = lane >> 4, n = lane & 15;
#pragma unroll
    for (int r8 = 0; r8 < 8; ++r8) {
      int m = r8 + kh * 8;
      int c0 = (wave * 2) * 16 + n;
      int c1 = (wave * 2 + 1) * 16 + n;
      l1buf[m * HID + c0] = fmaxf(acc0[r8] + b1[c0], 0.f);
      l1buf[m * HID + c1] = fmaxf(acc1[r8] + b1[c1], 0.f);
    }
  }
  __syncthreads();

  // l2: 256->1 per row
  {
    int row = tid >> 4, seg = tid & 15;
    float p = 0.f;
#pragma unroll
    for (int u = 0; u < 16; ++u) {
      int c = seg * 16 + u;
      p += l1buf[row * HID + c] * w2[c];
    }
    part[tid] = p;
  }
  __syncthreads();
  if (tid < 16) {
    float x = b2[0];
#pragma unroll
    for (int s = 0; s < 16; ++s) x += part[tid * 16 + s];
    int r = rowBase + tid;
    float lnum = 0.f, lden = 0.f;
    if (r < rows) {
      float nm = lmask[r];
      if (mode == 0) {
        float y = lab[r];
        float l = fmaxf(x, 0.f) - x * y + log1pf(__expf(-fabsf(x)));
        lnum = l * nm;
      } else {
        float d = x - lab[r];
        lnum = d * d * nm;
      }
      lden = nm;
    }
    rl[tid] = lnum;
    rd[tid] = lden;
  }
  __syncthreads();
  if (tid == 0) {
    float sn = 0.f, sd = 0.f;
#pragma unroll
    for (int s = 0; s < 16; ++s) { sn += rl[s]; sd += rd[s]; }
    atomicAdd(accN, sn);
    atomicAdd(accD, sd);
  }
}

__global__ void k_sloss(const float* __restrict__ h, const float* __restrict__ W,
                        const float* __restrict__ bias, const int* __restrict__ S,
                        const float* __restrict__ mask, float* __restrict__ acc) {
  int t = blockIdx.x * blockDim.x + threadIdx.x;
  if (t >= B_ * N_) return;
  float logit[VOCAB];
  float mx = -3.4e38f;
#pragma unroll 1
  for (int v = 0; v < VOCAB; ++v) {
    float s = bias[v];
    for (int c = 0; c < HID; ++c) s += h[(size_t)t * HID + c] * W[(size_t)c * VOCAB + v];
    logit[v] = s;
    mx = fmaxf(mx, s);
  }
  float se = 0.f;
  for (int v = 0; v < VOCAB; ++v) se += __expf(logit[v] - mx);
  float lse = logf(se) + mx;
  float sl = -(logit[S[t]] - lse);
  float m = mask[t];
  atomicAdd(&acc[ACC_S],     sl * m);
  atomicAdd(&acc[ACC_S + 1], m);
}

__global__ void k_final(const float* __restrict__ acc, float* __restrict__ out) {
  if (threadIdx.x == 0 && blockIdx.x == 0) {
    float e = 1e-12f;
    out[0] = acc[ACC_S] / (acc[ACC_S + 1] + e) + acc[ACC_N] / (acc[ACC_N + 1] + e) +
             acc[ACC_V] / (acc[ACC_V + 1] + e) + acc[ACC_D] / (acc[ACC_D + 1] + e);
  }
}

// ---------------------------------------------------------------- host
extern "C" void kernel_launch(void* const* d_in, const int* in_sizes, int n_in,
                              void* d_out, int out_size, void* d_ws, size_t ws_size,
                              hipStream_t stream) {
  (void)in_sizes; (void)n_in; (void)out_size; (void)ws_size;

  // ---- input mapping (jax pytree flatten order: dict keys sorted) ----
  const float* X     = (const float*)d_in[0];
  const float* noise = (const float*)d_in[1];
  const float* mask  = (const float*)d_in[2];
  const int*   S     = (const int*)d_in[3];
  // d_in[4] = L (unused)
  const float* Od_l1_b = (const float*)d_in[5];
  const float* Od_l1_w = (const float*)d_in[6];
  const float* Od_l2_b = (const float*)d_in[7];
  const float* Od_l2_w = (const float*)d_in[8];
  const float* On_l1_b = (const float*)d_in[9];
  const float* On_l1_w = (const float*)d_in[10];
  const float* On_l2_b = (const float*)d_in[11];
  const float* On_l2_w = (const float*)d_in[12];
  const float* Os_b    = (const float*)d_in[13];
  const float* Os_w    = (const float*)d_in[14];
  const float* Ov_b    = (const float*)d_in[15];
  const float* Ov_w    = (const float*)d_in[16];

  struct MpnW {
    const float *U_b[DEPTH], *U_w[DEPTH];
    const float *We_b, *We_w;
    const float *Wh_b[DEPTH], *Wh_w[DEPTH];
    const float *Ws;
    const float *Wv_b, *Wv_w;
  };
  auto load_mpn = [&](int base) {
    MpnW w;
    int i = base;
    for (int t = 0; t < DEPTH; ++t) { w.U_b[t] = (const float*)d_in[i++]; w.U_w[t] = (const float*)d_in[i++]; }
    w.We_b = (const float*)d_in[i++]; w.We_w = (const float*)d_in[i++];
    for (int t = 0; t < DEPTH; ++t) { w.Wh_b[t] = (const float*)d_in[i++]; w.Wh_w[t] = (const float*)d_in[i++]; }
    w.Ws = (const float*)d_in[i++];
    w.Wv_b = (const float*)d_in[i++]; w.Wv_w = (const float*)d_in[i++];
    return w;
  };
  MpnW Wseq    = load_mpn(17);
  MpnW Wstruct = load_mpn(38);

  // ---- workspace bump allocator ----
  char* wp = (char*)d_ws;
  auto alloc = [&](size_t bytes) -> void* {
    void* p = (void*)wp;
    wp += (bytes + 255) & ~(size_t)255;
    return p;
  };
  float* acc     = (float*)alloc(8 * sizeof(float));
  float* Xp      = (float*)alloc((size_t)B_ * N_ * 12 * sizeof(float));
  float* V       = (float*)alloc((size_t)B_ * N_ * 6 * sizeof(float));
  float* trueV   = (float*)alloc((size_t)B_ * N_ * 6 * sizeof(float));
  float* Dnei    = (float*)alloc((size_t)B_ * N_ * KNEI * sizeof(float));
  int*   Eidx    = (int*)alloc((size_t)B_ * N_ * KNEI * sizeof(int));
  int*   Etrue   = (int*)alloc((size_t)B_ * N_ * KNEI * sizeof(int));
  float* nlab    = (float*)alloc((size_t)B_ * (N_ - 1) * N_ * sizeof(float));
  float* nmask   = (float*)alloc((size_t)B_ * (N_ - 1) * N_ * sizeof(float));
  float* dlab    = (float*)alloc((size_t)B_ * (N_ - 1) * KNEI * sizeof(float));
  float* dmask   = (float*)alloc((size_t)B_ * (N_ - 1) * KNEI * sizeof(float));
  half_t* E16    = (half_t*)alloc((size_t)B_ * N_ * KNEI * EDGE_IN * 2);
  half_t* Av     = (half_t*)alloc((size_t)B_ * N_ * KPAD_WV * 2);
  half_t* Amsg   = (half_t*)alloc((size_t)B_ * N_ * KNEI * KPAD_MSG * 2);
  half_t* msg16  = (half_t*)alloc((size_t)B_ * N_ * KNEI * HID * 2);
  half_t* AU     = (half_t*)alloc((size_t)B_ * N_ * KPAD_MSG * 2);
  float*  hF     = (float*)alloc((size_t)B_ * N_ * HID * sizeof(float));
  half_t* hH     = (half_t*)alloc((size_t)B_ * N_ * HID * 2);
  half_t* he16   = (half_t*)alloc((size_t)B_ * N_ * KNEI * HID * 2);
  // swizzled weight fragments
  auto allocFrag = [&](int Kpad) { return (half_t*)alloc((size_t)Kpad * HID * 2); };
  half_t *frWv[2], *frWe[2], *frWh[2][DEPTH], *frU[2][DEPTH];
  for (int p = 0; p < 2; ++p) {
    frWv[p] = allocFrag(KPAD_WV);
    frWe[p] = allocFrag(EDGE_IN);
    for (int t = 0; t < DEPTH; ++t) { frWh[p][t] = allocFrag(KPAD_MSG); frU[p][t] = allocFrag(KPAD_MSG); }
  }
  half_t* frNei  = allocFrag(KPAD_NEI);
  half_t* frDist = allocFrag(KPAD_NEI);

  const int TB = 256;
  auto g1 = [&](int n) { return dim3((n + TB - 1) / TB); };

  // ---- pipeline ----
  k_zero<<<dim3(1), dim3(32), 0, stream>>>(acc, 8);
  k_axpy<<<g1(B_ * N_ * 12), dim3(TB), 0, stream>>>(X, noise, Xp, AUG_EPS, B_ * N_ * 12);
  k_dihedral<<<g1(B_ * 3 * N_), dim3(TB), 0, stream>>>(Xp, V);
  k_dihedral<<<g1(B_ * 3 * N_), dim3(TB), 0, stream>>>(X, trueV);
  k_knn<<<g1(B_ * N_), dim3(TB), 0, stream>>>(Xp, mask, Eidx, Dnei, 1);
  k_knn<<<g1(B_ * N_), dim3(TB), 0, stream>>>(X, mask, Etrue, nullptr, 0);
  k_edge<<<g1(B_ * N_ * KNEI), dim3(TB), 0, stream>>>(Eidx, Dnei, E16);
  k_neilabel<<<g1(B_ * (N_ - 1) * N_), dim3(TB), 0, stream>>>(Etrue, mask, X, nlab, nmask, dlab, dmask);

  // weight swizzles
  auto swz = [&](const float* W, half_t* out, int K, int Kpad) {
    int tot = (Kpad / 32) * NT * 32;
    k_swizzle<<<g1(tot), dim3(TB), 0, stream>>>(W, out, K, Kpad);
  };
  const MpnW* mp[2] = {&Wstruct, &Wseq};
  for (int p = 0; p < 2; ++p) {
    swz(mp[p]->Wv_w, frWv[p], NODE_IN + HID, KPAD_WV);
    swz(mp[p]->We_w, frWe[p], EDGE_IN, EDGE_IN);
    for (int t = 0; t < DEPTH; ++t) {
      swz(mp[p]->Wh_w[t], frWh[p][t], 2 * HID, KPAD_MSG);
      swz(mp[p]->U_w[t],  frU[p][t],  2 * HID, KPAD_MSG);
    }
  }
  swz(On_l1_w, frNei,  2 * HID + POS_DIM, KPAD_NEI);
  swz(Od_l1_w, frDist, 2 * HID + POS_DIM, KPAD_NEI);

  auto gemm = [&](const half_t* A, const half_t* Bf, const float* bias,
                  const float* rowm, float* Cf, half_t* Ch, int M, int Kpad) {
    k_gemm<<<dim3((M + 15) / 16), dim3(256), 0, stream>>>(A, Bf, bias, rowm, Cf, Ch, M, Kpad, 1);
  };

  auto run_mpn = [&](int p) {
    const MpnW& w = *mp[p];
    const int Mn = B_ * N_;
    const int Me = B_ * N_ * KNEI;
    k_fill_av<<<g1(Mn * KPAD_WV), dim3(TB), 0, stream>>>(V, w.Ws, S, Av);
    gemm(Av, frWv[p], w.Wv_b, mask, hF, hH, Mn, KPAD_WV);
    gemm(E16, frWe[p], w.We_b, nullptr, nullptr, he16, Me, EDGE_IN);
    for (int t = 0; t < DEPTH; ++t) {
      k_fill_amsg<<<g1(Me * 32), dim3(TB), 0, stream>>>(hH, he16, Eidx, Amsg);
      gemm(Amsg, frWh[p][t], w.Wh_b[t], nullptr, nullptr, msg16, Me, KPAD_MSG);
      k_reduce<<<g1(Mn * HID), dim3(TB), 0, stream>>>(msg16, Eidx, mask, hH, AU);
      gemm(AU, frU[p][t], w.U_b[t], mask, hF, hH, Mn, KPAD_MSG);
    }
  };

  // struct pass
  run_mpn(0);
  k_vloss<<<g1(B_ * (N_ - 1)), dim3(TB), 0, stream>>>(hF, Ov_w, Ov_b, trueV, mask, acc);

  // fused pairwise MLPs (use struct h in f16)
  {
    int rowsN = B_ * (N_ - 1) * N_;
    int rowsD = B_ * (N_ - 1) * KNEI;
    k_bigmlp<<<dim3((rowsN + 15) / 16), dim3(256), 0, stream>>>(
        hH, frNei, On_l1_b, On_l2_w, On_l2_b, nlab, nmask, &acc[ACC_N], &acc[ACC_N + 1], rowsN, 0);
    k_bigmlp<<<dim3((rowsD + 15) / 16), dim3(256), 0, stream>>>(
        hH, frDist, Od_l1_b, Od_l2_w, Od_l2_b, dlab, dmask, &acc[ACC_D], &acc[ACC_D + 1], rowsD, 1);
  }

  // seq pass (reuses h buffers)
  run_mpn(1);
  k_sloss<<<g1(B_ * N_), dim3(TB), 0, stream>>>(hF, Os_w, Os_b, S, mask, acc);

  k_final<<<dim3(1), dim3(32), 0, stream>>>(acc, (float*)d_out);
}
